// RNNModel_61512521613480
// MI455X (gfx1250) — matrix operand order repaired and checked by reading, not compiled, not run on hardware
//
#include <hip/hip_runtime.h>
#include <hip/hip_bf16.h>
#include <math.h>

// ---------------- problem constants ----------------
#define B_   64
#define T_   1024
#define U_   64
#define H_   512
#define M_   20
#define K_   10
#define A_   73
#define NKT  19             // gate-GEMM K padded to 608 = 19 chunks of 32
#define NCT  128            // 2048 / 16 column tiles
#define NAKT 16             // attention K = 512 = 16 chunks of 32
#define NWG  32
#define NTHR 256
#define EPSV   1e-8f
#define SIGEPS 1e-4f
#define TWO_PI 6.283185307179586f

typedef __attribute__((ext_vector_type(16))) _Float16 v16h;
typedef __attribute__((ext_vector_type(8)))  _Float16 v8h;
typedef __attribute__((ext_vector_type(8)))  float    v8f;

// ---------------- workspace layout (bytes) ----------------
#define WFRAG_HALVES (NCT * NKT * 32 * 16)            // 1,245,184 halves
#define AFRAG_HALVES (2 * NAKT * 32 * 16)             // 16,384 halves (Wattn, N padded to 32)
#define WS_WFRAG   0
#define WS_AFRAG   (WFRAG_HALVES * 2)
#define WS_HBUF    (WS_AFRAG + AFRAG_HALVES * 2)      // f32 h, double buffered [2][B][H]
#define WS_PSUM    (WS_HBUF   + 2 * B_ * H_ * 4)      // 64 partial sums
#define WS_PCNT    (WS_PSUM   + 64 * 4)               // 64 partial counts
#define WS_BAR     (WS_PCNT   + 64 * 4)               // barrier {cnt,gen}
#define WS_HBUF16  (WS_BAR    + 64)                   // f16 h, double buffered [2][B][H]
#define WS_TOTAL   (WS_HBUF16 + 2 * B_ * H_ * 2)

// ---------------- gate-weight fragment prep ----------------
// Wcat rows: 0..511 = Wh, 512..584 = Wx[3..75] (w part), 585..587 = Wx[0..2]
// (x part), 588..607 = zero pad.  Packed into WMMA B-fragment order:
//   frag[((ct*NKT + kt)*32 + lane)*16 + j] = Wcat[kt*32 + (lane>=16?8:0) + (j<8?j:8+j)][colmap]
// Column interleave col2 = unit*4 + gate  ->  original col = gate*512 + unit,
// so workgroup n owns global units [n*16, n*16+16) with i/f/g/o adjacent.
__global__ void prep_wfrag(const float* __restrict__ Wx, const float* __restrict__ Wh,
                           _Float16* __restrict__ frag) {
  int idx = blockIdx.x * blockDim.x + threadIdx.x;
  if (idx >= WFRAG_HALVES) return;
  int j    = idx & 15;
  int lane = (idx >> 4) & 31;
  int tile = idx >> 9;
  int kt   = tile % NKT;
  int ct   = tile / NKT;
  int Kg   = kt * 32 + ((lane & 16) ? 8 : 0) + ((j < 8) ? j : 8 + j);
  int col2 = ct * 16 + (lane & 15);
  int unit = col2 >> 2, gate = col2 & 3;
  int ocol = gate * H_ + unit;
  float v = 0.f;
  if      (Kg < 512) v = Wh[Kg * 2048 + ocol];
  else if (Kg < 585) v = Wx[(3 + Kg - 512) * 2048 + ocol];
  else if (Kg < 588) v = Wx[(Kg - 585) * 2048 + ocol];
  frag[idx] = (_Float16)v;
}

// Wattn [512][30] -> B-fragments for a 64x32 (N padded) x K=512 WMMA GEMM.
__global__ void prep_afrag(const float* __restrict__ Wattn, _Float16* __restrict__ frag) {
  int idx = blockIdx.x * blockDim.x + threadIdx.x;
  if (idx >= AFRAG_HALVES) return;
  int j    = idx & 15;
  int lane = (idx >> 4) & 31;
  int tile = idx >> 9;
  int kt   = tile % NAKT;
  int ct   = tile / NAKT;
  int Kg   = kt * 32 + ((lane & 16) ? 8 : 0) + ((j < 8) ? j : 8 + j);
  int col  = ct * 16 + (lane & 15);
  frag[idx] = (_Float16)((col < 30) ? Wattn[Kg * 30 + col] : 0.f);
}

__global__ void init_state(int* __restrict__ p, int n) {
  int i = blockIdx.x * blockDim.x + threadIdx.x;
  if (i < n) p[i] = 0;
}

// ---------------- grid-wide sense barrier (persistent grid) ----------------
__device__ __forceinline__ void grid_barrier(int* bar) {
  __threadfence();
  __syncthreads();
  if (threadIdx.x == 0) {
    int g = __hip_atomic_load(bar + 1, __ATOMIC_RELAXED, __HIP_MEMORY_SCOPE_AGENT);
    int a = __hip_atomic_fetch_add(bar, 1, __ATOMIC_ACQ_REL, __HIP_MEMORY_SCOPE_AGENT);
    if (a == NWG - 1) {
      __hip_atomic_store(bar, 0, __ATOMIC_RELAXED, __HIP_MEMORY_SCOPE_AGENT);
      __hip_atomic_fetch_add(bar + 1, 1, __ATOMIC_RELEASE, __HIP_MEMORY_SCOPE_AGENT);
    } else {
      while (__hip_atomic_load(bar + 1, __ATOMIC_ACQUIRE, __HIP_MEMORY_SCOPE_AGENT) == g)
        __builtin_amdgcn_s_sleep(1);
    }
  }
  __syncthreads();
}

// ---------------- persistent recurrent kernel ----------------
// One grid barrier per timestep.  Attention is replicated in every WG (state in
// LDS, deterministic), so w_t never crosses workgroups.
__global__ __launch_bounds__(NTHR, 1) void rnn_persistent(
    const float* __restrict__ x,      const float* __restrict__ y,
    const float* __restrict__ b_lstm, const float* __restrict__ b_attn,
    const float* __restrict__ Wg,     const float* __restrict__ b_g,
    const int* __restrict__ x_len,    const int* __restrict__ c,
    const int* __restrict__ c_len,    char* __restrict__ ws) {
  const _Float16* __restrict__ wfrag = (const _Float16*)(ws + WS_WFRAG);
  const _Float16* __restrict__ afrag = (const _Float16*)(ws + WS_AFRAG);
  float*    hbuf   = (float*)(ws + WS_HBUF);      // [2][B][H]
  _Float16* hbuf16 = (_Float16*)(ws + WS_HBUF16); // [2][B][H]
  float*    psum   = (float*)(ws + WS_PSUM);
  int*      pcnt   = (int*)(ws + WS_PCNT);
  int*      bar    = (int*)(ws + WS_BAR);

  __shared__ float gbuf[B_][66];   // gate tile after cross-lane regroup   16,896 B
  __shared__ float ab_s[B_][32];   // attention alpha/beta/kappa scratch    8,192 B
  __shared__ float phi_s[B_][64];  // attention weights                    16,384 B
  __shared__ float w_s[B_][76];    // soft window w_t (73 used)            19,456 B
  __shared__ float kap_s[B_][K_];  // persistent kappa state                2,560 B
  __shared__ float z_s[2][128];    // GMM head outputs for 2 batches        1,024 B
  __shared__ float cst_s[1];       // (unused pad)

  const int tid  = threadIdx.x;
  const int wg   = blockIdx.x;
  const int wv   = tid >> 5;            // wave32
  const int lane = tid & 31;
  const int bg   = wv & 3;              // batch-group 0..3 (16 rows each)
  const int cg0  = (wv >> 2) * 2;       // 2 col-groups per wave (gate GEMM)
  const int act  = wv >> 2;             // attention col-tile 0..1
  const int row  = bg * 16 + (lane & 15);
  const int hi   = (lane & 16) ? 8 : 0; // A-fragment / D-row half select

  // per-WG LSTM cell state for the 16 units this WG owns: 64 batches x 16 units
  // = 1024 values -> 4 per thread in registers.
  float creg[4] = {0.f, 0.f, 0.f, 0.f};

  // init LDS attention state
  for (int i = tid; i < B_ * 76; i += NTHR) w_s[i / 76][i % 76] = 0.f;
  for (int i = tid; i < B_ * K_; i += NTHR) kap_s[i / K_][i % K_] = 0.f;
  __syncthreads();

  float accS = 0.f;
  int   accN = 0;

  for (int t = 0; t < T_; ++t) {
    const int rd = t & 1, wr = (t + 1) & 1;
    const _Float16* __restrict__ h16 = hbuf16 + (size_t)rd * B_ * H_;

    // ---- gate GEMM: [64 x 608] x [608 x 64] per WG, WMMA f16->f32 ----
    v8f d0 = {}; v8f d1 = {};
    const _Float16* bp0 = wfrag + ((size_t)(wg * 4 + cg0)     * NKT) * 512 + lane * 16;
    const _Float16* bp1 = wfrag + ((size_t)(wg * 4 + cg0 + 1) * NKT) * 512 + lane * 16;
    for (int kt = 0; kt < NKT; ++kt) {
      v16h a;
      const int base = kt * 32 + hi;    // K start for this lane half
      if (kt < 16) {                    // pure-h chunks: direct f16 loads
        const v8h* hp = (const v8h*)(h16 + row * H_ + base);
        v8h lo = hp[0];                 // K = base..base+7
        v8h hiq = hp[2];                // K = base+16..base+23
        a = __builtin_shufflevector(lo, hiq, 0,1,2,3,4,5,6,7,8,9,10,11,12,13,14,15);
      } else {                          // w / x / pad tail chunks (w from LDS)
        #pragma unroll
        for (int j = 0; j < 16; ++j) {
          int Kg = base + ((j < 8) ? j : 8 + j);
          float v = 0.f;
          if      (Kg < 585) v = (Kg >= 512) ? w_s[row][Kg - 512] : 0.f;
          else if (Kg < 588) v = x[((size_t)row * T_ + t) * 3 + (Kg - 585)];
          a[j] = (_Float16)v;
        }
      }
      v16h b0 = *(const v16h*)(bp0 + (size_t)kt * 512);
      v16h b1 = *(const v16h*)(bp1 + (size_t)kt * 512);
      d0 = __builtin_amdgcn_wmma_f32_16x16x32_f16(false, a, false, b0, (short)0, d0, false, false);
      d1 = __builtin_amdgcn_wmma_f32_16x16x32_f16(false, a, false, b1, (short)0, d1, false, false);
    }
    // D layout: VGPR r holds M=r (lanes 0-15) / M=r+8 (lanes 16-31), N=lane&15
    #pragma unroll
    for (int r = 0; r < 8; ++r) {
      int br = bg * 16 + hi + r;
      gbuf[br][cg0 * 16 + (lane & 15)]       = d0[r];
      gbuf[br][(cg0 + 1) * 16 + (lane & 15)] = d1[r];
    }
    __syncthreads();

    // ---- LSTM pointwise: this WG owns global units [wg*16, wg*16+16) ----
    #pragma unroll
    for (int q = 0; q < 4; ++q) {
      int idx = tid + q * NTHR;
      int b = idx >> 4, lu = idx & 15;
      int gu = wg * 16 + lu;
      float gi = gbuf[b][lu * 4 + 0] + b_lstm[gu];
      float gf = gbuf[b][lu * 4 + 1] + b_lstm[512 + gu];
      float gg = gbuf[b][lu * 4 + 2] + b_lstm[1024 + gu];
      float go = gbuf[b][lu * 4 + 3] + b_lstm[1536 + gu];
      float si = 1.f / (1.f + expf(-gi));
      float sf = 1.f / (1.f + expf(-gf));
      float so = 1.f / (1.f + expf(-go));
      float cc = sf * creg[q] + si * tanhf(gg);
      creg[q] = cc;
      float h = so * tanhf(cc);
      hbuf[(size_t)wr * B_ * H_ + b * H_ + gu]   = h;
      hbuf16[(size_t)wr * B_ * H_ + b * H_ + gu] = (_Float16)h;
    }
    grid_barrier(bar);                  // h_t complete & visible everywhere

    const float*    __restrict__ hcur = hbuf   + (size_t)wr * B_ * H_;
    const _Float16* __restrict__ hc16 = hbuf16 + (size_t)wr * B_ * H_;

    // ---- attention (replicated in every WG): ab = h @ Wattn via WMMA ----
    {
      v8f av = {};
      const _Float16* ap = afrag + ((size_t)act * NAKT) * 512 + lane * 16;
      for (int kt = 0; kt < NAKT; ++kt) {
        const v8h* hp = (const v8h*)(hc16 + row * H_ + kt * 32 + hi);
        v16h a = __builtin_shufflevector(hp[0], hp[2],
                                         0,1,2,3,4,5,6,7,8,9,10,11,12,13,14,15);
        v16h b0 = *(const v16h*)(ap + (size_t)kt * 512);
        av = __builtin_amdgcn_wmma_f32_16x16x32_f16(false, a, false, b0, (short)0, av, false, false);
      }
      int col = act * 16 + (lane & 15);
      float bias = (col < 30) ? b_attn[col] : 0.f;
      #pragma unroll
      for (int r = 0; r < 8; ++r)
        ab_s[bg * 16 + hi + r][col] = av[r] + bias;
    }
    __syncthreads();
    // kappa update (alpha/beta/kappa transformed in place)
    for (int idx = tid; idx < B_ * K_; idx += NTHR) {
      int b = idx / K_, kk = idx % K_;
      float al = expf(ab_s[b][kk]);
      float be = expf(ab_s[b][10 + kk]);
      float kp = kap_s[b][kk] + expf(ab_s[b][20 + kk]);
      kap_s[b][kk] = kp;
      ab_s[b][kk] = al; ab_s[b][10 + kk] = be; ab_s[b][20 + kk] = kp;
    }
    __syncthreads();
    // phi over U
    for (int idx = tid; idx < B_ * U_; idx += NTHR) {
      int b = idx >> 6, u = idx & 63;
      float s = 0.f;
      if (u < c_len[b]) {
        float uf = (float)u;
        #pragma unroll
        for (int kk = 0; kk < K_; ++kk) {
          float dd = ab_s[b][20 + kk] - uf;
          s += ab_s[b][kk] * expf(-ab_s[b][10 + kk] * dd * dd);
        }
      }
      phi_s[b][u] = s;
    }
    __syncthreads();
    for (int i = tid; i < B_ * 76; i += NTHR) w_s[i / 76][i % 76] = 0.f;
    __syncthreads();
    if (tid < B_) {
      int b = tid;
      for (int u = 0; u < U_; ++u)
        w_s[b][c[b * U_ + u]] += phi_s[b][u];
    }

    // ---- fused GMM-NLL head for this step: 2 batches per WG ----
    for (int idx = tid; idx < 2 * 121; idx += NTHR) {
      int bi = idx / 121, j = idx % 121;
      int b = wg * 2 + bi;
      float s = b_g[j];
      for (int k = 0; k < H_; ++k) s += hcur[b * H_ + k] * Wg[k * 121 + j];
      z_s[bi][j] = s;
    }
    __syncthreads();                    // also orders w_s scatter before next GEMM
    if (tid < 2) {
      int b = wg * 2 + tid;
      const float* z = z_s[tid];
      float mx = z[0];
      for (int m = 1; m < M_; ++m) mx = fmaxf(mx, z[m]);
      float se = 0.f;
      for (int m = 0; m < M_; ++m) se += expf(z[m] - mx);
      float y1 = y[((size_t)b * T_ + t) * 3 + 0];
      float y2 = y[((size_t)b * T_ + t) * 3 + 1];
      float y3 = y[((size_t)b * T_ + t) * 3 + 2];
      float gmm = 0.f;
      for (int m = 0; m < M_; ++m) {
        float pi  = expf(z[m] - mx) / se;
        float s1  = fmaxf(expf(z[20 + m]), SIGEPS);
        float s2  = fmaxf(expf(z[40 + m]), SIGEPS);
        float rho = fminf(fmaxf(tanhf(z[60 + m]), EPSV - 1.f), 1.f - EPSV);
        float d1v = (y1 - z[80 + m]) / s1;
        float d2v = (y2 - z[100 + m]) / s2;
        float omr = 1.f - rho * rho;
        float Zq  = d1v * d1v + d2v * d2v - 2.f * rho * d1v * d2v;
        float nrm = 1.f / (TWO_PI * s1 * s2 * sqrtf(omr));
        gmm += pi * expf(-Zq / (2.f * omr)) * nrm;
      }
      gmm = fmaxf(gmm, EPSV);
      float es = 1.f / (1.f + expf(-z[120]));
      es = fminf(fmaxf(es, EPSV), 1.f - EPSV);
      float bern = (y3 == 1.f) ? es : (1.f - es);
      float nll = -(logf(gmm) + logf(bern));
      if ((t < x_len[b]) && !__builtin_isnan(nll)) { accS += nll; accN++; }
    }
  }

  if (tid < 2) { psum[wg * 2 + tid] = accS; pcnt[wg * 2 + tid] = accN; }
  (void)cst_s;
}

// ---------------- deterministic final reduction ----------------
__global__ void finalize_loss(const float* __restrict__ psum,
                              const int* __restrict__ pcnt,
                              float* __restrict__ out) {
  float s = 0.f; int n = 0;
  for (int i = 0; i < 64; ++i) { s += psum[i]; n += pcnt[i]; }
  out[0] = s / fmaxf((float)n, 1.f);
}

extern "C" void kernel_launch(void* const* d_in, const int* in_sizes, int n_in,
                              void* d_out, int out_size, void* d_ws, size_t ws_size,
                              hipStream_t stream) {
  (void)in_sizes; (void)n_in; (void)out_size; (void)ws_size;
  const float* x      = (const float*)d_in[0];
  const float* y      = (const float*)d_in[1];
  const float* Wx     = (const float*)d_in[2];
  const float* Wh     = (const float*)d_in[3];
  const float* b_lstm = (const float*)d_in[4];
  const float* Wattn  = (const float*)d_in[5];
  const float* b_attn = (const float*)d_in[6];
  const float* Wg     = (const float*)d_in[7];
  const float* b_g    = (const float*)d_in[8];
  const int*   x_len  = (const int*)d_in[9];
  const int*   c      = (const int*)d_in[10];
  const int*   c_len  = (const int*)d_in[11];
  char* ws = (char*)d_ws;

  prep_wfrag<<<(WFRAG_HALVES + 255) / 256, 256, 0, stream>>>(
      Wx, Wh, (_Float16*)(ws + WS_WFRAG));
  prep_afrag<<<(AFRAG_HALVES + 255) / 256, 256, 0, stream>>>(
      Wattn, (_Float16*)(ws + WS_AFRAG));
  int zero_dwords = (WS_TOTAL - WS_HBUF) / 4;
  init_state<<<(zero_dwords + 255) / 256, 256, 0, stream>>>(
      (int*)(ws + WS_HBUF), zero_dwords);
  rnn_persistent<<<NWG, NTHR, 0, stream>>>(
      x, y, b_lstm, b_attn, Wg, b_g, x_len, c, c_len, ws);
  finalize_loss<<<1, 1, 0, stream>>>(
      (const float*)(ws + WS_PSUM), (const int*)(ws + WS_PCNT), (float*)d_out);
}
